// SimilarityLoss_43147241455942
// MI455X (gfx1250) — matrix operand order repaired and checked by reading, not compiled, not run on hardware
//
#include <hip/hip_runtime.h>
#include <hip/hip_bf16.h>
#include <math.h>

// Problem constants from the reference
#define BB     128
#define NWAY   5
#define QRY    15
#define NSLOT  32
#define HH     256
#define NQ     (NWAY * QRY)          // 75 queries per batch
#define NSUP   (BB * NWAY)           // 640 support rows
#define NG     (BB * NQ)             // 9600 (b,i) pairs
#define LAMBDA_F 1.0f

typedef __attribute__((ext_vector_type(16))) _Float16 v16h;
typedef __attribute__((ext_vector_type(8)))  float    v8f;

// ---------------------------------------------------------------------------
// Kernel A: scores[b*5+w][s] = mean_h sup[b,w,s,h].  One wave per (b,w,s).
// ---------------------------------------------------------------------------
__global__ __launch_bounds__(256) void kA_scores(const float* __restrict__ out_f,
                                                 float* __restrict__ scores) {
    int lane = threadIdx.x & 31;
    int task = blockIdx.x * 8 + (threadIdx.x >> 5);   // 0..20479 == (b*5+w)*32+s
    const float* p = out_f + (size_t)task * HH;       // support rows are first 640
    float s = 0.0f;
#pragma unroll
    for (int j = 0; j < 8; ++j) s += p[lane + 32 * j];
#pragma unroll
    for (int off = 16; off; off >>= 1) s += __shfl_xor(s, off);
    if (lane == 0) scores[task] = s * (1.0f / (float)HH);
}

// ---------------------------------------------------------------------------
// Kernel B: greedy slot selection, one wave per batch (lane == slot).
// Ties broken toward the lower slot index (matches stable argsort + argmax).
// ---------------------------------------------------------------------------
__global__ __launch_bounds__(32) void kB_select(const float* __restrict__ scores,
                                                int* __restrict__ sel) {
    int b = blockIdx.x;
    int s = threadIdx.x;
    bool used = false;
    for (int j = 0; j < NWAY; ++j) {
        float v = used ? -3.402823466e38f : scores[(b * NWAY + j) * NSLOT + s];
        int idx = s;
#pragma unroll
        for (int off = 16; off; off >>= 1) {
            float ov = __shfl_xor(v, off);
            int   oi = __shfl_xor(idx, off);
            if (ov > v || (ov == v && oi < idx)) { v = ov; idx = oi; }
        }
        if (s == idx) used = true;
        if (s == 0) sel[b * NWAY + j] = idx;
    }
}

// ---------------------------------------------------------------------------
// Kernel C: proto[(b*5+k)][0:256] = normalize(sup[b,k,sel[b,k],:]).
// One wave per (b,k).
// ---------------------------------------------------------------------------
__global__ __launch_bounds__(256) void kC_proto(const float* __restrict__ out_f,
                                                const int* __restrict__ sel,
                                                float* __restrict__ proto) {
    int lane = threadIdx.x & 31;
    int task = blockIdx.x * 8 + (threadIdx.x >> 5);   // b*5+k, 0..639
    int slot = sel[task];
    const float* src = out_f + ((size_t)task * NSLOT + (size_t)slot) * HH;
    float x[8];
    float ss = 0.0f;
#pragma unroll
    for (int j = 0; j < 8; ++j) { x[j] = src[lane + 32 * j]; ss += x[j] * x[j]; }
#pragma unroll
    for (int off = 16; off; off >>= 1) ss += __shfl_xor(ss, off);
    float inv = 1.0f / fmaxf(sqrtf(ss), 1e-12f);
#pragma unroll
    for (int j = 0; j < 8; ++j) proto[(size_t)task * HH + lane + 32 * j] = x[j] * inv;
}

// ---------------------------------------------------------------------------
// Kernel D: WMMA dot products.  One wave per (b,k,tile); tile = 16 query rows.
// A (16x32 f16): lane L<16 holds row L, K = h0+{0..7,16..23}; lane L+16 holds
//                row L, K = h0+{8..15,24..31} (ISA 16-bit A layout).
// B (32x16 f16): lane L holds column L&15; lanes 0-15 -> K=h0+{0..7,16..23},
//                lanes 16-31 -> K=h0+{8..15,24..31}.  Only column 0 = proto.
// D (16x16 f32): column 0 is in lanes 0 (M=0..7) and 16 (M=8..15).
// diff[b,i,k] = 2 - 2 * (row_i . proto_k) / max(||row_i||, 1e-12).
// ---------------------------------------------------------------------------
__global__ __launch_bounds__(32) void kD_dots(const float* __restrict__ out_f,
                                              const int* __restrict__ sel,
                                              const float* __restrict__ proto,
                                              float* __restrict__ diff) {
    __shared__ float lds[32];                 // [0..15] dots, [16..31] norms^2

    int task = blockIdx.x;                    // 0..3199
    int tile = task % 5;
    int bk   = task / 5;                      // b*5+k
    int b = bk / NWAY;
    int k = bk - b * NWAY;
    int lane = threadIdx.x;
    int rowL = lane & 15;
    int half = lane >> 4;

    int slot = sel[bk];
    int i  = tile * 16 + rowL;
    int ic = (i < NQ) ? i : (NQ - 1);         // clamp padding rows (ignored later)
    int w = ic / QRY;
    int q = ic - w * QRY;
    // query row (640 + b*75 + w*15 + q), slot `slot`
    const float* rowptr = out_f +
        ((size_t)(NSUP + b * NQ + w * QRY + q) * (NSLOT * HH) + (size_t)slot * HH);
    const float* pp = proto + (size_t)bk * HH;
    float scaleB = (rowL == 0) ? 1.0f : 0.0f; // only B column 0 carries proto

    v8f c = {};
    float nsq = 0.0f;
#pragma unroll
    for (int ch = 0; ch < 8; ++ch) {
        int h0 = ch * 32;
        int baseA = h0 + half * 8;
        float4 a0 = *(const float4*)(rowptr + baseA);
        float4 a1 = *(const float4*)(rowptr + baseA + 4);
        float4 a2 = *(const float4*)(rowptr + baseA + 16);
        float4 a3 = *(const float4*)(rowptr + baseA + 20);
        nsq += a0.x*a0.x + a0.y*a0.y + a0.z*a0.z + a0.w*a0.w
             + a1.x*a1.x + a1.y*a1.y + a1.z*a1.z + a1.w*a1.w
             + a2.x*a2.x + a2.y*a2.y + a2.z*a2.z + a2.w*a2.w
             + a3.x*a3.x + a3.y*a3.y + a3.z*a3.z + a3.w*a3.w;
        v16h a;
        a[0]=(_Float16)a0.x;  a[1]=(_Float16)a0.y;  a[2]=(_Float16)a0.z;  a[3]=(_Float16)a0.w;
        a[4]=(_Float16)a1.x;  a[5]=(_Float16)a1.y;  a[6]=(_Float16)a1.z;  a[7]=(_Float16)a1.w;
        a[8]=(_Float16)a2.x;  a[9]=(_Float16)a2.y;  a[10]=(_Float16)a2.z; a[11]=(_Float16)a2.w;
        a[12]=(_Float16)a3.x; a[13]=(_Float16)a3.y; a[14]=(_Float16)a3.z; a[15]=(_Float16)a3.w;

        int baseB = h0 + half * 8;
        float4 b0 = *(const float4*)(pp + baseB);
        float4 b1 = *(const float4*)(pp + baseB + 4);
        float4 b2 = *(const float4*)(pp + baseB + 16);
        float4 b3 = *(const float4*)(pp + baseB + 20);
        v16h bb;
        bb[0]=(_Float16)(b0.x*scaleB);  bb[1]=(_Float16)(b0.y*scaleB);
        bb[2]=(_Float16)(b0.z*scaleB);  bb[3]=(_Float16)(b0.w*scaleB);
        bb[4]=(_Float16)(b1.x*scaleB);  bb[5]=(_Float16)(b1.y*scaleB);
        bb[6]=(_Float16)(b1.z*scaleB);  bb[7]=(_Float16)(b1.w*scaleB);
        bb[8]=(_Float16)(b2.x*scaleB);  bb[9]=(_Float16)(b2.y*scaleB);
        bb[10]=(_Float16)(b2.z*scaleB); bb[11]=(_Float16)(b2.w*scaleB);
        bb[12]=(_Float16)(b3.x*scaleB); bb[13]=(_Float16)(b3.y*scaleB);
        bb[14]=(_Float16)(b3.z*scaleB); bb[15]=(_Float16)(b3.w*scaleB);

        c = __builtin_amdgcn_wmma_f32_16x16x32_f16(
                /*neg_a=*/false, a, /*neg_b=*/false, bb,
                /*c_mod=*/(short)0, c, /*reuse_a=*/false, /*reuse_b=*/false);
    }

    // row norm^2: lanes L and L+16 together cover all K of row L
    float tot = nsq + __shfl_xor(nsq, 16);
    if (lane < 16) lds[16 + lane] = tot;
    if (lane == 0) {
#pragma unroll
        for (int m = 0; m < 8; ++m) lds[m] = c[m];         // D[0..7, 0]
    } else if (lane == 16) {
#pragma unroll
        for (int m = 0; m < 8; ++m) lds[8 + m] = c[m];     // D[8..15, 0]
    }
    __syncthreads();
    if (lane < 16) {
        int i2 = tile * 16 + lane;
        if (i2 < NQ) {
            float nrm = sqrtf(lds[16 + lane]);
            float dt  = lds[lane] / fmaxf(nrm, 1e-12f);
            diff[((size_t)b * NQ + i2) * NWAY + k] = 2.0f - 2.0f * dt;
        }
    }
}

// ---------------------------------------------------------------------------
// Kernel E: per (b,i): nll = diff[label] + logsumexp(-diff); correct flag.
// Flat label index == flat (b,i) index (the 75-dim reshape is a pure view).
// ---------------------------------------------------------------------------
__global__ __launch_bounds__(256) void kE_nll(const float* __restrict__ diff,
                                              const int* __restrict__ labels,
                                              float* __restrict__ nllArr,
                                              float* __restrict__ corArr) {
    int g = blockIdx.x * 256 + threadIdx.x;
    if (g >= NG) return;
    float d[NWAY];
#pragma unroll
    for (int k = 0; k < NWAY; ++k) d[k] = diff[(size_t)g * NWAY + k];
    float mn = d[0];
    int   am = 0;
#pragma unroll
    for (int k = 1; k < NWAY; ++k)
        if (d[k] < mn) { mn = d[k]; am = k; }   // strict < == first-occurrence argmax of logits
    float sum = 0.0f;
#pragma unroll
    for (int k = 0; k < NWAY; ++k) sum += expf(-(d[k] - mn));
    float lse = -mn + logf(sum);                 // logsumexp over -d
    int lab = labels[g];
    nllArr[g] = d[lab] + lse;
    corArr[g] = (am == lab) ? 1.0f : 0.0f;
}

// ---------------------------------------------------------------------------
// Kernel F: deterministic fixed-order reduction + finalize.
// ---------------------------------------------------------------------------
__global__ __launch_bounds__(256) void kF_final(const float* __restrict__ nllArr,
                                                const float* __restrict__ corArr,
                                                const float* __restrict__ att_loss,
                                                float* __restrict__ outp) {
    __shared__ float s1[256];
    __shared__ float s2[256];
    int t = threadIdx.x;
    float a = 0.0f, cc = 0.0f;
    for (int g = t; g < NG; g += 256) { a += nllArr[g]; cc += corArr[g]; }
    s1[t] = a; s2[t] = cc;
    __syncthreads();
    for (int off = 128; off; off >>= 1) {
        if (t < off) { s1[t] += s1[t + off]; s2[t] += s2[t + off]; }
        __syncthreads();
    }
    if (t == 0) {
        outp[0] = s1[0] / (float)NG + LAMBDA_F * att_loss[0];
        outp[1] = s2[0] / (float)NG;
    }
}

// ---------------------------------------------------------------------------
// Workspace (floats): scores 20480 | proto 163840 | diff 48000 | nll 9600 |
// cor 9600 | sel 640 ints  -> ~1.01 MB total.
// ---------------------------------------------------------------------------
extern "C" void kernel_launch(void* const* d_in, const int* in_sizes, int n_in,
                              void* d_out, int out_size, void* d_ws, size_t ws_size,
                              hipStream_t stream) {
    (void)in_sizes; (void)n_in; (void)out_size; (void)ws_size;
    const float* out_f    = (const float*)d_in[0];
    // d_in[1] = labels_support (unused in forward)
    const int*   labels_q = (const int*)d_in[2];
    const float* att_loss = (const float*)d_in[3];
    // d_in[4] = mode (unused)

    float* ws     = (float*)d_ws;
    float* scores = ws;                 // 20480
    float* proto  = ws + 20480;         // 163840
    float* diffv  = ws + 184320;        // 48000
    float* nllArr = ws + 232320;        // 9600
    float* corArr = ws + 241920;        // 9600
    int*   sel    = (int*)(ws + 251520);// 640 ints

    kA_scores<<<2560, 256, 0, stream>>>(out_f, scores);
    kB_select<<<BB, 32, 0, stream>>>(scores, sel);
    kC_proto <<<80, 256, 0, stream>>>(out_f, sel, proto);
    kD_dots  <<<NSUP * 5, 32, 0, stream>>>(out_f, sel, proto, diffv);
    kE_nll   <<<(NG + 255) / 256, 256, 0, stream>>>(diffv, labels_q, nllArr, corArr);
    kF_final <<<1, 256, 0, stream>>>(nllArr, corArr, att_loss, (float*)d_out);
}
